// VPTBlock_22359599743444
// MI455X (gfx1250) — matrix operand order repaired and checked by reading, not compiled, not run on hardware
//
#include <hip/hip_runtime.h>
#include <hip/hip_bf16.h>
#include <math.h>

typedef _Float16 f16;
typedef __attribute__((ext_vector_type(16))) _Float16 v16h;
typedef __attribute__((ext_vector_type(8)))  _Float16 v8h;
typedef __attribute__((ext_vector_type(8)))  float    v8f;

#define C_DIM   768
#define NHEAD   12
#define HD      64
#define WSZ     14
#define VPT_    10
#define BATCH   8
#define H0_     64
#define NWH     5          // windows per side (70/14)
#define NWIN    25
#define BW      200        // NWIN * BATCH
#define LW      206        // VPT + WSZ*WSZ
#define LPAD    224        // LW padded to 7*32 (attention K padding)
#define L_TOK   4106       // VPT + H0*H0
#define M_QKV   (BW*LW)    // 41200
#define NHEADS_ (BW*NHEAD) // 2400
#define HSTRIDE LPAD       // per-head row stride of q/k/v
#define SCALE_  0.125f
#define M_MLP   (BATCH*L_TOK)  // 32848
#define M_CHUNK 8212           // M_MLP/4
#define QKV_ELE ((size_t)NHEADS_*HSTRIDE*HD)  // 34,406,400 elems per tensor

// ---------------- helpers ----------------
__device__ __forceinline__ float block_reduce256(float v, float* red) {
  int tid = threadIdx.x;
  red[tid] = v; __syncthreads();
  for (int s = 128; s > 0; s >>= 1) {
    if (tid < s) red[tid] += red[tid + s];
    __syncthreads();
  }
  float r = red[0]; __syncthreads();
  return r;
}

// CDNA5 async global->LDS copy, 16B per lane, tracked by ASYNCcnt.
__device__ __forceinline__ void async_ld_b128(uint32_t lds_addr, const void* gaddr) {
  asm volatile("global_load_async_to_lds_b128 %0, %1, off"
               :: "v"(lds_addr), "v"(gaddr) : "memory");
}
__device__ __forceinline__ void wait_asynccnt0() {
  asm volatile("s_wait_asynccnt 0" ::: "memory");
}
__device__ __forceinline__ uint32_t lds_addr32(const void* p) {
  // generic pointer to __shared__ has the LDS byte offset in its low 32 bits
  return (uint32_t)(uintptr_t)p;
}

__global__ void cvt_f16_k(const float* __restrict__ in, f16* __restrict__ out, int n) {
  for (int i = blockIdx.x * blockDim.x + threadIdx.x; i < n; i += gridDim.x * blockDim.x)
    out[i] = (f16)in[i];
}

__global__ void copy_f32_k(const float* __restrict__ in, float* __restrict__ out, size_t n) {
  for (size_t i = blockIdx.x * (size_t)blockDim.x + threadIdx.x; i < n; i += (size_t)gridDim.x * blockDim.x)
    out[i] = in[i];
}

// ---------------- LayerNorm1 + window partition -> tok f16 ----------------
__global__ __launch_bounds__(256) void ln1_tok_k(const float* __restrict__ x,
                                                 const float* __restrict__ w,
                                                 const float* __restrict__ b,
                                                 f16* __restrict__ tok) {
  __shared__ float red[256];
  int t = blockIdx.x;            // 0..M_QKV-1
  int bw = t / LW, l = t % LW;
  const float* src = nullptr;
  if (l < VPT_) {
    int bb = bw % BATCH;                         // prompt broadcast ordering (win*B + b)
    src = x + ((size_t)bb * L_TOK + l) * C_DIM;
  } else {
    int bb  = bw / NWIN;                         // window ordering (b*25 + win)
    int win = bw % NWIN;
    int pos = l - VPT_;
    int h = (win / NWH) * WSZ + pos / WSZ;
    int wc = (win % NWH) * WSZ + pos % WSZ;
    if (h < H0_ && wc < H0_)
      src = x + ((size_t)bb * L_TOK + VPT_ + h * H0_ + wc) * C_DIM;
  }
  f16* dst = tok + (size_t)t * C_DIM;
  int tid = threadIdx.x;
  if (!src) {                                    // zero-padded window position
    for (int c = tid; c < C_DIM; c += 256) dst[c] = (f16)0.0f;
    return;
  }
  float s = 0.f, sq = 0.f, vals[3];
  for (int i = 0; i < 3; ++i) { float v = src[tid + i*256]; vals[i] = v; s += v; sq += v*v; }
  s  = block_reduce256(s,  red);
  sq = block_reduce256(sq, red);
  float mean = s  * (1.0f / C_DIM);
  float var  = sq * (1.0f / C_DIM) - mean * mean;
  float rstd = rsqrtf(var + 1e-5f);
  for (int i = 0; i < 3; ++i) {
    int c = tid + i*256;
    dst[c] = (f16)((vals[i] - mean) * rstd * w[c] + b[c]);
  }
}

// ---------------- generic row LayerNorm -> f16 ----------------
__global__ __launch_bounds__(256) void ln_rows_k(const float* __restrict__ in,
                                                 const float* __restrict__ w,
                                                 const float* __restrict__ b,
                                                 f16* __restrict__ out) {
  __shared__ float red[256];
  const float* src = in + (size_t)blockIdx.x * C_DIM;
  f16* dst = out + (size_t)blockIdx.x * C_DIM;
  int tid = threadIdx.x;
  float s = 0.f, sq = 0.f, vals[3];
  for (int i = 0; i < 3; ++i) { float v = src[tid + i*256]; vals[i] = v; s += v; sq += v*v; }
  s  = block_reduce256(s,  red);
  sq = block_reduce256(sq, red);
  float mean = s  * (1.0f / C_DIM);
  float var  = sq * (1.0f / C_DIM) - mean * mean;
  float rstd = rsqrtf(var + 1e-5f);
  for (int i = 0; i < 3; ++i) {
    int c = tid + i*256;
    dst[c] = (f16)((vals[i] - mean) * rstd * w[c] + b[c]);
  }
}

// ---------------- rel-pos bias: rel_h/rel_w einsums ----------------
__global__ __launch_bounds__(256) void relpos_k(const f16* __restrict__ q,
                                                const float* __restrict__ rph,
                                                const float* __restrict__ rpw,
                                                float* __restrict__ relh,
                                                float* __restrict__ relw) {
  int bh = blockIdx.x;
  const f16* qb = q + (size_t)bh * HSTRIDE * HD;
  for (int e = threadIdx.x; e < WSZ*WSZ*WSZ; e += 256) {
    int h = e / (WSZ*WSZ);
    int rem = e % (WSZ*WSZ);
    int w = rem / WSZ;
    int k = rem % WSZ;
    const f16* qr = qb + (size_t)(VPT_ + h*WSZ + w) * HD;
    const float* rh = rph + (h - k + WSZ - 1) * HD;
    const float* rw = rpw + (w - k + WSZ - 1) * HD;
    float sh = 0.f, sw = 0.f;
    for (int c = 0; c < HD; ++c) { float qv = (float)qr[c]; sh += qv*rh[c]; sw += qv*rw[c]; }
    size_t o = (size_t)bh * (WSZ*WSZ*WSZ) + e;
    relh[o] = sh; relw[o] = sw;
  }
}

// ---------------- tiled WMMA GEMM: D = A(MxK,f16) * B(NxK,f16)^T + bias ----------------
// 128x64 block tile, BK=64, double-buffered LDS filled by async global->LDS copies.
// EPI 0: QKV scatter  1: proj scatter+atomic  2: GELU->f16  3: residual add -> f32
#define BKD 64

template<int EPI>
__global__ __launch_bounds__(256) void gemm_f16_k(const f16* __restrict__ A,
                                                  const f16* __restrict__ Bm,
                                                  const float* __restrict__ bias,
                                                  int M, int N, int K,
                                                  f16* __restrict__ o16,
                                                  float* __restrict__ o32,
                                                  const float* __restrict__ aux) {
  __shared__ f16 As[2][128*BKD];   // 2 x 16 KB
  __shared__ f16 Bs[2][64*BKD];    // 2 x  8 KB
  const int tid  = threadIdx.x;
  const int lane = tid & 31, wave = tid >> 5;
  const int wm = wave & 3, wn = wave >> 2;       // 4x2 wave grid -> 128x64 tile
  const int lmod = lane & 15, lhal = lane >> 4;
  const int bm0 = blockIdx.x * 128, bn0 = blockIdx.y * 64;

  // per-thread staging assignment (addresses advance by 64 halves each K-step)
  const int arow = tid >> 1, acb = (tid & 1) * 32;        // A: 32 halves (64B) per thread
  const int brow = tid >> 2, bcb = (tid & 3) * 16;        // B: 16 halves (32B) per thread
  const f16* gA = A + (size_t)(bm0 + arow) * K + acb;     // unguarded: OOB rows land in ws, discarded
  const f16* gB = Bm + (size_t)(bn0 + brow) * K + bcb;

  const int nk = K / BKD;

  // prefetch first K-tile into buffer 0
  {
    uint32_t la = lds_addr32(&As[0][arow*BKD + acb]);
    for (int j = 0; j < 4; ++j) async_ld_b128(la + j*16, (const char*)gA + j*16);
    uint32_t lb = lds_addr32(&Bs[0][brow*BKD + bcb]);
    for (int j = 0; j < 2; ++j) async_ld_b128(lb + j*16, (const char*)gB + j*16);
  }

  v8f acc[2][2] = {};
  for (int ks = 0; ks < nk; ++ks) {
    wait_asynccnt0();          // our async fills (issued last iter) are resident
    __syncthreads();           // everyone done reading the buffer we are about to refill
    int cur = ks & 1;
    if (ks + 1 < nk) {         // kick off next K-tile into the other buffer
      int nxt = cur ^ 1;
      const char* pa = (const char*)(gA + (size_t)(ks + 1) * BKD);
      uint32_t la = lds_addr32(&As[nxt][arow*BKD + acb]);
      for (int j = 0; j < 4; ++j) async_ld_b128(la + j*16, pa + j*16);
      const char* pb = (const char*)(gB + (size_t)(ks + 1) * BKD);
      uint32_t lb = lds_addr32(&Bs[nxt][brow*BKD + bcb]);
      for (int j = 0; j < 2; ++j) async_ld_b128(lb + j*16, pb + j*16);
    }
    // compute on current buffer: two 32-wide sub-steps
    for (int kc = 0; kc < 2; ++kc) {
      v16h af[2], bf[2];
      for (int ti = 0; ti < 2; ++ti) {         // A frag: row=lmod, K split by lhal
        const f16* ar = &As[cur][(wm*32 + ti*16 + lmod)*BKD + kc*32 + lhal*16];
        for (int j = 0; j < 8; ++j) { af[ti][j] = ar[j]; af[ti][8+j] = ar[8+j]; }
      }
      for (int tj = 0; tj < 2; ++tj) {         // B frag: col=lmod, K=lhal*16+i
        const f16* br = &Bs[cur][(wn*32 + tj*16 + lmod)*BKD + kc*32 + lhal*16];
        for (int i = 0; i < 16; ++i) bf[tj][i] = br[i];
      }
      for (int ti = 0; ti < 2; ++ti)
        for (int tj = 0; tj < 2; ++tj)
          acc[ti][tj] = __builtin_amdgcn_wmma_f32_16x16x32_f16(
              false, af[ti], false, bf[tj], (short)0, acc[ti][tj], false, false);
    }
  }
  // epilogue
  for (int ti = 0; ti < 2; ++ti)
  for (int tj = 0; tj < 2; ++tj)
  for (int j = 0; j < 8; ++j) {
    int gm = bm0 + wm*32 + ti*16 + j + 8*lhal;
    int gn = bn0 + wn*32 + tj*16 + lmod;
    if (gm >= M) continue;
    float val = acc[ti][tj][j] + bias[gn];
    if (EPI == 0) {                 // QKV: scatter to per-head q/k/v (stride-224 rows)
      int part = gn / C_DIM, rest = gn % C_DIM;
      int head = rest / HD, d = rest % HD;
      int bw = gm / LW, l = gm % LW;
      int bh = bw * NHEAD + head;
      f16* dst = o16 + (size_t)part * QKV_ELE;
      dst[((size_t)bh * HSTRIDE + l) * HD + d] = (f16)val;
    } else if (EPI == 1) {          // proj: window re-assembly + prompt mean into x2
      int bw = gm / LW, l = gm % LW;
      if (l < VPT_) {
        int b = bw % BATCH;
        atomicAdd(o32 + ((size_t)b * L_TOK + l) * C_DIM + gn, val * 0.04f); // mean over 25 windows
      } else {
        int b = bw / NWIN, win = bw % NWIN;
        int pos = l - VPT_;
        int h = (win / NWH) * WSZ + pos / WSZ;
        int w = (win % NWH) * WSZ + pos % WSZ;
        if (h < H0_ && w < H0_)
          atomicAdd(o32 + ((size_t)b * L_TOK + VPT_ + h * H0_ + w) * C_DIM + gn, val);
      }
    } else if (EPI == 2) {          // exact GELU -> f16
      float g = 0.5f * val * (1.0f + erff(val * 0.70710678118f));
      o16[(size_t)gm * N + gn] = (f16)g;
    } else {                        // MLP2: out = x2 + val
      size_t idx = (size_t)gm * N + gn;
      o32[idx] = aux[idx] + val;
    }
  }
}

// ---------------- attention: per wave one head; S=q*k^T*SCALE+rel, softmax, O=P*v ----------------
__global__ __launch_bounds__(64) void attn_k(const f16* __restrict__ q,
                                             const f16* __restrict__ k,
                                             const f16* __restrict__ v,
                                             const float* __restrict__ relh,
                                             const float* __restrict__ relw,
                                             f16* __restrict__ out) {
  __shared__ float S[2*16*LPAD];
  __shared__ f16   P[2*16*LPAD];
  const int lane = threadIdx.x & 31, wave = threadIdx.x >> 5;
  const int bh = blockIdx.x * 2 + wave;
  const int bw = bh / NHEAD, head = bh % NHEAD;
  const int lmod = lane & 15, lhal = lane >> 4;
  float* Sw = S + wave * 16 * LPAD;
  f16*   Pw = P + wave * 16 * LPAD;
  const f16* qb = q + (size_t)bh * HSTRIDE * HD;
  const f16* kb = k + (size_t)bh * HSTRIDE * HD;
  const f16* vb = v + (size_t)bh * HSTRIDE * HD;
  const float* rhb = relh + (size_t)bh * (WSZ*WSZ*WSZ);
  const float* rwb = relw + (size_t)bh * (WSZ*WSZ*WSZ);

  for (int rt = 0; rt < 13; ++rt) {            // 13 row tiles cover 208 >= 206 rows
    int r0 = rt * 16;
    v16h a0, a1;                               // q rows r0..r0+15, K=0..63 (two K=32 frags)
    {
      const f16* qr = qb + (size_t)(r0 + lmod) * HD + lhal * 16;
      for (int j = 0; j < 8; ++j) {
        a0[j] = qr[j];    a0[8+j] = qr[8+j];
        a1[j] = qr[32+j]; a1[8+j] = qr[40+j];
      }
    }
    for (int ct = 0; ct < 14; ++ct) {          // 14 col tiles cover LPAD=224
      int c0 = ct * 16;
      v16h b0, b1;
      const f16* kr = kb + (size_t)(c0 + lmod) * HD + lhal * 16;
      for (int i = 0; i < 16; ++i) { b0[i] = kr[i]; b1[i] = kr[32+i]; }
      v8f acc = {};
      acc = __builtin_amdgcn_wmma_f32_16x16x32_f16(false, a0, false, b0, (short)0, acc, false, false);
      acc = __builtin_amdgcn_wmma_f32_16x16x32_f16(false, a1, false, b1, (short)0, acc, false, false);
      for (int j = 0; j < 8; ++j)
        Sw[(j + 8*lhal) * LPAD + c0 + lmod] = acc[j] * SCALE_;
    }
    __syncthreads();
    // decomposed rel-pos bias + pad-column mask
    for (int e = lane; e < 16*LPAD; e += 32) {
      int r = e / LPAD, c = e % LPAD;
      int gr = r0 + r;
      if (c >= LW) Sw[e] = -1e30f;
      else if (gr < LW && gr >= VPT_ && c >= VPT_) {
        int pq = gr - VPT_, pk = c - VPT_;
        int hk = pk / WSZ, wk = pk % WSZ;
        Sw[e] += rhb[pq * WSZ + hk] + rwb[pq * WSZ + wk];
      }
    }
    __syncthreads();
    // softmax per row (lanes 0..15 each own one row), P in f16, pad cols/rows -> 0
    if (lane < 16) {
      int gr = r0 + lane;
      float* sr = Sw + lane * LPAD;
      f16*   pr = Pw + lane * LPAD;
      if (gr < LW) {
        float m = -1e30f;
        for (int c = 0; c < LW; ++c) m = fmaxf(m, sr[c]);
        float sum = 0.f;
        for (int c = 0; c < LW; ++c) { float ev = __expf(sr[c] - m); sr[c] = ev; sum += ev; }
        float inv = 1.0f / sum;
        for (int c = 0; c < LW; ++c)    pr[c] = (f16)(sr[c] * inv);
        for (int c = LW; c < LPAD; ++c) pr[c] = (f16)0.0f;
      } else {
        for (int c = 0; c < LPAD; ++c)  pr[c] = (f16)0.0f;
      }
    }
    __syncthreads();
    // O(16x64) = P(16x224) @ V(224x64)
    for (int nt = 0; nt < 4; ++nt) {
      int n0 = nt * 16;
      v8f acc = {};
      for (int kt = 0; kt < 7; ++kt) {
        int kk = kt * 32;
        v16h ap, bv;
        const f16* prow = Pw + lmod * LPAD + kk + lhal * 16;
        for (int j = 0; j < 8; ++j) { ap[j] = prow[j]; ap[8+j] = prow[8+j]; }
        const f16* vr = vb + (size_t)(kk + lhal * 16) * HD + n0 + lmod;
        for (int i = 0; i < 16; ++i) bv[i] = vr[(size_t)i * HD];
        acc = __builtin_amdgcn_wmma_f32_16x16x32_f16(false, ap, false, bv, (short)0, acc, false, false);
      }
      for (int j = 0; j < 8; ++j) {
        int gr = r0 + j + 8*lhal;
        if (gr < LW)
          out[((size_t)(bw * LW + gr)) * C_DIM + head * HD + n0 + lmod] = (f16)acc[j];
      }
    }
    __syncthreads();
  }
}

// ---------------- host ----------------
extern "C" void kernel_launch(void* const* d_in, const int* in_sizes, int n_in,
                              void* d_out, int out_size, void* d_ws, size_t ws_size,
                              hipStream_t stream) {
  (void)in_sizes; (void)n_in; (void)out_size; (void)ws_size;
  const float* x      = (const float*)d_in[0];
  const float* ln1_w  = (const float*)d_in[1];
  const float* ln1_b  = (const float*)d_in[2];
  const float* qkv_w  = (const float*)d_in[3];
  const float* qkv_b  = (const float*)d_in[4];
  const float* proj_w = (const float*)d_in[5];
  const float* proj_b = (const float*)d_in[6];
  const float* rph    = (const float*)d_in[7];
  const float* rpw    = (const float*)d_in[8];
  const float* ln2_w  = (const float*)d_in[9];
  const float* ln2_b  = (const float*)d_in[10];
  const float* mlp_w1 = (const float*)d_in[11];
  const float* mlp_b1 = (const float*)d_in[12];
  const float* mlp_w2 = (const float*)d_in[13];
  const float* mlp_b2 = (const float*)d_in[14];
  float* outp = (float*)d_out;

  char* ws = (char*)d_ws;
  const size_t MB = 1ull << 20;
  f16*   w_qkv  = (f16*)(ws + 0*MB);       // 3.4 MB
  f16*   w_proj = (f16*)(ws + 4*MB);       // 1.2 MB
  f16*   w_mlp1 = (f16*)(ws + 6*MB);       // 4.5 MB
  f16*   w_mlp2 = (f16*)(ws + 11*MB);      // 4.5 MB
  f16*   tok    = (f16*)(ws + 16*MB);      // 63.3 MB; reused as attn_out
  f16*   qkv    = (f16*)(ws + 80*MB);      // 3 x 65.6 MB (q,k,v)
  float* x2     = (float*)(ws + 80*MB);    // 96.3 MB, overlays dead q/k after attention
  f16*   xn2    = (f16*)(ws + 184*MB);     // 48.2 MB, overlays dead k/v
  f16*   hbuf   = (f16*)(ws + 236*MB);     // 48.2 MB, overlays dead v tail
  float* relh   = (float*)(ws + 288*MB);   // 25.2 MB
  float* relw   = (float*)(ws + 316*MB);   // 25.2 MB  (peak ws ~342 MB)

  // 1) weights -> f16
  cvt_f16_k<<<1024, 256, 0, stream>>>(qkv_w,  w_qkv,  3*C_DIM*C_DIM);
  cvt_f16_k<<<1024, 256, 0, stream>>>(proj_w, w_proj, C_DIM*C_DIM);
  cvt_f16_k<<<1024, 256, 0, stream>>>(mlp_w1, w_mlp1, 4*C_DIM*C_DIM);
  cvt_f16_k<<<1024, 256, 0, stream>>>(mlp_w2, w_mlp2, 4*C_DIM*C_DIM);

  // 2) LN1 + window partition -> tok
  ln1_tok_k<<<M_QKV, 256, 0, stream>>>(x, ln1_w, ln1_b, tok);

  // 3) QKV GEMM (zero q/k/v first so padded rows are clean for WMMA)
  hipMemsetAsync(qkv, 0, 3 * QKV_ELE * sizeof(f16), stream);
  gemm_f16_k<0><<<dim3((M_QKV + 127)/128, 2304/64), 256, 0, stream>>>(
      tok, w_qkv, qkv_b, M_QKV, 2304, C_DIM, qkv, nullptr, nullptr);

  // 4) rel-pos bias tables
  relpos_k<<<NHEADS_, 256, 0, stream>>>(qkv, rph, rpw, relh, relw);

  // 5) attention (writes attn_out into tok region)
  attn_k<<<NHEADS_/2, 64, 0, stream>>>(qkv, qkv + QKV_ELE, qkv + 2*QKV_ELE,
                                       relh, relw, tok);

  // 6) x2 = shortcut; proj GEMM scatters (+mean of prompt rows) into x2
  copy_f32_k<<<4096, 256, 0, stream>>>(x, x2, (size_t)M_MLP * C_DIM);
  gemm_f16_k<1><<<dim3((M_QKV + 127)/128, C_DIM/64), 256, 0, stream>>>(
      tok, w_proj, proj_b, M_QKV, C_DIM, C_DIM, nullptr, x2, nullptr);

  // 7) LN2 -> f16
  ln_rows_k<<<M_MLP, 256, 0, stream>>>(x2, ln2_w, ln2_b, xn2);

  // 8) MLP in 4 row-chunks (reuse hbuf): h = gelu(xn2 @ W1^T + b1); out = x2 + h @ W2^T + b2
  for (int c = 0; c < 4; ++c) {
    size_t ro = (size_t)c * M_CHUNK;
    gemm_f16_k<2><<<dim3((M_CHUNK + 127)/128, (4*C_DIM)/64), 256, 0, stream>>>(
        xn2 + ro * C_DIM, w_mlp1, mlp_b1, M_CHUNK, 4*C_DIM, C_DIM, hbuf, nullptr, nullptr);
    gemm_f16_k<3><<<dim3((M_CHUNK + 127)/128, C_DIM/64), 256, 0, stream>>>(
        hbuf, w_mlp2, mlp_b2, M_CHUNK, C_DIM, 4*C_DIM, nullptr,
        outp + ro * C_DIM, x2 + ro * C_DIM);
  }
}
